// GCN_12987981103642
// MI455X (gfx1250) — matrix operand order repaired and checked by reading, not compiled, not run on hardware
//
#include <hip/hip_runtime.h>

// ---------------------------------------------------------------------------
// GCN pipeline for MI455X (gfx1250, wave32, WMMA).
// GEMMs (conv2 implicit-im2col + ARMA dense) run on v_wmma_f32_16x16x32_bf16
// with f32 accumulation; graph scatter uses global f32 atomics.
// Block tile 64x64 (8 wave32, each wave 16x32 = 2 WMMAs/K-step).
// Tile staging: batched unconditional clamped loads (full MLP), mask-by-mul,
// then convert+store to LDS.
// ---------------------------------------------------------------------------

typedef __attribute__((ext_vector_type(16))) __bf16 v16bf;
typedef __attribute__((ext_vector_type(8)))  float  v8f;

#define N_NODES 5000
#define N_EDGES 160000
#define CHUNK   250   // node chunk for CNN encoder (bounds workspace)

__device__ __forceinline__ float leaky01(float v) { return v > 0.f ? v : 0.01f * v; }

// ---------------------------------------------------------------------------
// WMMA GEMM: C[M,N] = A[M,K] * B[K,N] (+bias, leaky if EPI==1)
// AMODE 0: A is row-major f32 [M,K]
// AMODE 1: implicit im2col for conv2: A points at conv1out [cn][100][51][4],
//          m -> (node, oh(49), ow(3)); k -> (cin(100), kh(3), kw(2))
// ---------------------------------------------------------------------------
template<int AMODE, int EPI>
__global__ void __launch_bounds__(256) wmma_gemm(
    const float* __restrict__ A, const float* __restrict__ B,
    float* __restrict__ C, const float* __restrict__ bias,
    int M, int N, int K)
{
  __shared__ __bf16 As[64][40];   // [m][k] (+pad)
  __shared__ __bf16 Bs[64][40];   // [n][k] transposed (+pad)

  const int tid  = threadIdx.x;
  const int lane = tid & 31;
  const int wv   = tid >> 5;          // wave 0..7
  const int lo   = lane & 15;
  const int hi   = lane >> 4;
  const int mSub = (wv & 3) * 16;     // 4 waves across M
  const int nSub = (wv >> 2) * 32;    // 2 waves across N, 32 wide each
  const int mBase = blockIdx.y * 64;
  const int nBase = blockIdx.x * 64;

  v8f acc0 = {}, acc1 = {};

  // A staging: thread -> (row, 8 consecutive k)
  const int arow  = tid >> 2;         // 0..63
  const int akq   = (tid & 3) * 8;    // 0/8/16/24
  const int gm    = mBase + arow;
  const bool rowOk = gm < M;
  const int gmc   = rowOk ? gm : 0;
  int imBase = 0;                     // im2col thread-invariant base offset
  if (AMODE == 1) {
    int node = gmc / 147;
    int rem  = gmc - node * 147;
    int oh   = rem / 3;
    int ow   = rem - oh * 3;
    imBase   = node * 20400 + oh * 4 + ow;   // node*(100*51*4) + oh*4 + ow
  }
  // B staging: thread -> (k row, 8 consecutive n)
  const int brow = tid >> 3;          // 0..31
  const int bnq  = (tid & 7) * 8;     // 0..56

  const bool kAligned = ((K & 7) == 0);
  const bool nVec     = ((N & 3) == 0);

  for (int kb = 0; kb < K; kb += 32) {
    // ---- stage A tile (64m x 32k): phase-1 gather, phase-2 convert+store ----
    float va[8];
    if (AMODE == 0) {
      if (rowOk && kAligned && (kb + akq < K)) {
        // two b128 loads: K%8==0 -> 8-group fully in range, 16B aligned
        const float4* p = (const float4*)(A + (long)gm * K + (kb + akq));
        float4 a0 = p[0];
        float4 a1 = p[1];
        va[0] = a0.x; va[1] = a0.y; va[2] = a0.z; va[3] = a0.w;
        va[4] = a1.x; va[5] = a1.y; va[6] = a1.z; va[7] = a1.w;
      } else {
        #pragma unroll
        for (int i = 0; i < 8; ++i) {
          int kk  = kb + akq + i;
          int kkc = kk < K ? kk : K - 1;
          va[i] = A[(long)gmc * K + kkc];
        }
      }
    } else {
      #pragma unroll
      for (int i = 0; i < 8; ++i) {
        int kk  = kb + akq + i;
        int kkc = kk < K ? kk : K - 1;
        int cin = (int)((unsigned)kkc / 6u);
        int r   = kkc - cin * 6;
        va[i] = A[imBase + cin * 204 + (r >> 1) * 4 + (r & 1)];
      }
    }
    #pragma unroll
    for (int i = 0; i < 8; ++i) {
      int kk  = kb + akq + i;
      float m = (rowOk && kk < K) ? 1.f : 0.f;   // mask-by-mul: no exec diamonds
      As[arow][akq + i] = (__bf16)(va[i] * m);
    }

    // ---- stage B tile (32k x 64n) transposed: Bs[n][k] ----
    {
      int kk   = kb + brow;
      bool kOk = kk < K;
      int kkc  = kOk ? kk : K - 1;
      const float* bp = B + (long)kkc * N;
      float vb[8];
      int gn0 = nBase + bnq;
      if (nVec && (gn0 + 8 <= N)) {
        float4 b0 = *(const float4*)(bp + gn0);       // 16B aligned: N%4==0, gn0%4==0
        float4 b1 = *(const float4*)(bp + gn0 + 4);
        vb[0] = b0.x; vb[1] = b0.y; vb[2] = b0.z; vb[3] = b0.w;
        vb[4] = b1.x; vb[5] = b1.y; vb[6] = b1.z; vb[7] = b1.w;
      } else {
        #pragma unroll
        for (int i = 0; i < 8; ++i) {
          int gn  = gn0 + i;
          int gnc = gn < N ? gn : N - 1;
          vb[i] = bp[gnc];
        }
      }
      float km = kOk ? 1.f : 0.f;
      #pragma unroll
      for (int i = 0; i < 8; ++i) {
        float m = ((gn0 + i) < N) ? km : 0.f;
        Bs[bnq + i][brow] = (__bf16)(vb[i] * m);
      }
    }
    __syncthreads();

    // ---- build fragments, contiguous k = 16*laneHi + s for A and B, then 2 WMMAs ----
    v16bf af, bf0, bf1;
    #pragma unroll
    for (int v = 0; v < 8; ++v) {
      int j  = 2 * v;
      int ka = j + 16 * hi;               // A: k = j + 16*laneHi
      af[j]      = As[mSub + lo][ka];
      af[j + 1]  = As[mSub + lo][ka + 1];
      int kbx = j + 16 * hi;              // B: k = j + 16*laneHi
      bf0[j]     = Bs[nSub + lo][kbx];
      bf0[j + 1] = Bs[nSub + lo][kbx + 1];
      bf1[j]     = Bs[nSub + 16 + lo][kbx];
      bf1[j + 1] = Bs[nSub + 16 + lo][kbx + 1];
    }
    acc0 = __builtin_amdgcn_wmma_f32_16x16x32_bf16(false, af, false, bf0,
                                                   (short)0, acc0, false, false);
    acc1 = __builtin_amdgcn_wmma_f32_16x16x32_bf16(false, af, false, bf1,
                                                   (short)0, acc1, false, false);
    __syncthreads();
  }

  // ---- epilogue: C VGPR r -> m = r + 8*laneHi, n = lane&15 ----
  #pragma unroll
  for (int r = 0; r < 8; ++r) {
    int om  = mBase + mSub + r + 8 * hi;
    int on0 = nBase + nSub + lo;
    int on1 = on0 + 16;
    if (om < M) {
      if (on0 < N) {
        float v = acc0[r];
        if (EPI == 1) v = leaky01(v + bias[on0]);
        C[(long)om * N + on0] = v;
      }
      if (on1 < N) {
        float v = acc1[r];
        if (EPI == 1) v = leaky01(v + bias[on1]);
        C[(long)om * N + on1] = v;
      }
    }
  }
}

// ---------------------------------------------------------------------------
// Conv1: [cn,1,53,6] -> [cn,100,51,4], leaky relu, 3x3 VALID
// ---------------------------------------------------------------------------
__global__ void conv1_k(const float* __restrict__ x, const float* __restrict__ w,
                        const float* __restrict__ b, float* __restrict__ out,
                        int node0, int cn)
{
  int idx = blockIdx.x * blockDim.x + threadIdx.x;
  int total = cn * 100 * 51 * 4;
  if (idx >= total) return;
  int ow = idx & 3;
  int t = idx >> 2;
  int oh = t % 51; t /= 51;
  int co = t % 100;
  int nl = t / 100;
  const float* ip = x + (long)(node0 + nl) * (53 * 6);
  const float* wp = w + co * 9;
  float acc = b[co];
  #pragma unroll
  for (int kh = 0; kh < 3; ++kh)
    #pragma unroll
    for (int kw = 0; kw < 3; ++kw)
      acc += ip[(oh + kh) * 6 + (ow + kw)] * wp[kh * 3 + kw];
  out[idx] = leaky01(acc);
}

// ---------------------------------------------------------------------------
// Conv3: conv2out [cn,49,3,200] -> h94 [N_NODES,94], tanh.
// Weights (200*3*2=1200 f32) staged in LDS once per block.
// ---------------------------------------------------------------------------
__global__ void __launch_bounds__(256) conv3_k(
    const float* __restrict__ c2, const float* __restrict__ w,
    const float* __restrict__ b, float* __restrict__ h94,
    int node0, int cn)
{
  __shared__ float ws3[1200];
  for (int i = threadIdx.x; i < 1200; i += 256) ws3[i] = w[i];
  __syncthreads();

  int idx = blockIdx.x * blockDim.x + threadIdx.x;
  int total = cn * 94;
  if (idx >= total) return;
  int p  = idx % 94;
  int nl = idx / 94;
  int oh2 = p >> 1, ow2 = p & 1;
  float acc = b[0];
  for (int kh = 0; kh < 3; ++kh)
    for (int kw = 0; kw < 2; ++kw) {
      const float* cp = c2 + ((long)(nl * 49 + oh2 + kh) * 3 + (ow2 + kw)) * 200;
      const float* wp = ws3 + kh * 2 + kw;
      for (int ci = 0; ci < 200; ++ci)
        acc += cp[ci] * wp[ci * 6];
    }
  h94[(long)(node0 + nl) * 94 + p] = tanhf(acc);
}

// conv2 weight OIHW [200][600] -> [600][200] (GEMM B layout)
__global__ void transpose_w2(const float* __restrict__ w, float* __restrict__ wt)
{
  int idx = blockIdx.x * blockDim.x + threadIdx.x;
  if (idx >= 200 * 600) return;
  int co = idx / 600, kk = idx - co * 600;
  wt[kk * 200 + co] = w[idx];
}

// ---------------------------------------------------------------------------
// Graph normalization
// ---------------------------------------------------------------------------
__global__ void deg_k(const int* __restrict__ dst, float* __restrict__ deg, int E)
{
  int e = blockIdx.x * blockDim.x + threadIdx.x;
  if (e >= E) return;
  atomicAdd(&deg[dst[e]], 1.0f);
}

__global__ void dinv_k(float* __restrict__ deg, int n)   // in-place deg -> d^{-1/2}
{
  int i = blockIdx.x * blockDim.x + threadIdx.x;
  if (i >= n) return;
  float d = deg[i];
  deg[i] = d > 0.f ? rsqrtf(fmaxf(d, 1.f)) : 0.f;
}

__global__ void norm_k(const int* __restrict__ src, const int* __restrict__ dst,
                       const float* __restrict__ dinv, float* __restrict__ nrm, int E)
{
  int e = blockIdx.x * blockDim.x + threadIdx.x;
  if (e >= E) return;
  nrm[e] = dinv[src[e]] * dinv[dst[e]];
}

// ---------------------------------------------------------------------------
// ARMA edge scatter: agg[k][dst] += hinit[k][src] * norm
// ---------------------------------------------------------------------------
__global__ void scatter_k(const float* __restrict__ hinit, const int* __restrict__ src,
                          const int* __restrict__ dst, const float* __restrict__ nrm,
                          float* __restrict__ agg, int E, int F, int K)
{
  int idx = blockIdx.x * blockDim.x + threadIdx.x;
  if (idx >= E * F) return;
  int e = idx / F, f = idx - (idx / F) * F;
  int s = src[e], d = dst[e];
  float nm = nrm[e];
  for (int k = 0; k < K; ++k)
    atomicAdd(&agg[((long)k * N_NODES + d) * F + f],
              hinit[((long)k * N_NODES + s) * F + f] * nm);
}

// out = mean_k relu(agg + root + bias), optional tanh
__global__ void combine_k(const float* __restrict__ agg, const float* __restrict__ root,
                          const float* __restrict__ bias, float* __restrict__ out,
                          int F, int K, int doTanh)
{
  int idx = blockIdx.x * blockDim.x + threadIdx.x;
  if (idx >= N_NODES * F) return;
  int n = idx / F, f = idx - n * F;
  float s = 0.f;
  for (int k = 0; k < K; ++k) {
    float v = agg[((long)k * N_NODES + n) * F + f]
            + root[((long)k * N_NODES + n) * F + f]
            + bias[k * F + f];
    s += v > 0.f ? v : 0.f;
  }
  s *= (1.0f / (float)K);
  out[idx] = doTanh ? tanhf(s) : s;
}

__global__ void softmax_k(const float* __restrict__ in, float* __restrict__ out,
                          int n, int F)
{
  int i = blockIdx.x * blockDim.x + threadIdx.x;
  if (i >= n) return;
  const float* r = in + (long)i * F;
  float m = r[0];
  for (int f = 1; f < F; ++f) m = fmaxf(m, r[f]);
  float e[16], s = 0.f;
  for (int f = 0; f < F; ++f) { e[f] = __expf(r[f] - m); s += e[f]; }
  float inv = 1.f / s;
  for (int f = 0; f < F; ++f) out[(long)i * F + f] = e[f] * inv;
}

// ---------------------------------------------------------------------------
extern "C" void kernel_launch(void* const* d_in, const int* in_sizes, int n_in,
                              void* d_out, int out_size, void* d_ws, size_t ws_size,
                              hipStream_t stream)
{
  (void)in_sizes; (void)n_in; (void)out_size; (void)ws_size;

  const float* x   = (const float*)d_in[0];
  const int*   ei  = (const int*)d_in[1];
  const float* e1w = (const float*)d_in[2];
  const float* e1b = (const float*)d_in[3];
  const float* e2w = (const float*)d_in[4];
  const float* e2b = (const float*)d_in[5];
  const float* e3w = (const float*)d_in[6];
  const float* e3b = (const float*)d_in[7];
  const float *awi[6], *awr[6], *ab[6];
  for (int i = 0; i < 6; ++i) {
    awi[i] = (const float*)d_in[8 + 3 * i];
    awr[i] = (const float*)d_in[9 + 3 * i];
    ab[i]  = (const float*)d_in[10 + 3 * i];
  }
  const int* src = ei;
  const int* dst = ei + N_EDGES;

  // workspace carve-out (~73 MB)
  char* base = (char*)d_ws;
  size_t off = 0;
  auto alloc = [&](size_t bytes) -> void* {
    void* p = base + off;
    off += (bytes + 255) & ~(size_t)255;
    return p;
  };
  float* conv1out = (float*)alloc((size_t)CHUNK * 100 * 51 * 4 * 4);
  float* conv2out = (float*)alloc((size_t)CHUNK * 147 * 200 * 4);
  float* wt2      = (float*)alloc((size_t)600 * 200 * 4);
  float* h94      = (float*)alloc((size_t)N_NODES * 94 * 4);
  float* hA       = (float*)alloc((size_t)N_NODES * 128 * 4);
  float* hB       = (float*)alloc((size_t)N_NODES * 128 * 4);
  float* hinit    = (float*)alloc((size_t)2 * N_NODES * 128 * 4);
  float* root     = (float*)alloc((size_t)2 * N_NODES * 128 * 4);
  float* agg      = (float*)alloc((size_t)2 * N_NODES * 128 * 4);
  float* deg      = (float*)alloc((size_t)N_NODES * 4);
  float* nrm      = (float*)alloc((size_t)N_EDGES * 4);

  // ---- gcn_norm ----
  hipMemsetAsync(deg, 0, (size_t)N_NODES * 4, stream);
  deg_k <<<(N_EDGES + 255) / 256, 256, 0, stream>>>(dst, deg, N_EDGES);
  dinv_k<<<(N_NODES + 255) / 256, 256, 0, stream>>>(deg, N_NODES);
  norm_k<<<(N_EDGES + 255) / 256, 256, 0, stream>>>(src, dst, deg, nrm, N_EDGES);

  // ---- CNN encoder (chunked over nodes) ----
  transpose_w2<<<(200 * 600 + 255) / 256, 256, 0, stream>>>(e2w, wt2);
  for (int c0 = 0; c0 < N_NODES; c0 += CHUNK) {
    int cn = (N_NODES - c0 < CHUNK) ? (N_NODES - c0) : CHUNK;
    int t1 = cn * 100 * 51 * 4;
    conv1_k<<<(t1 + 255) / 256, 256, 0, stream>>>(x, e1w, e1b, conv1out, c0, cn);
    int M2 = cn * 147;
    dim3 g2((200 + 63) / 64, (M2 + 63) / 64);
    wmma_gemm<1, 1><<<g2, 256, 0, stream>>>(conv1out, wt2, conv2out, e2b, M2, 200, 600);
    int t3 = cn * 94;
    conv3_k<<<(t3 + 255) / 256, 256, 0, stream>>>(conv2out, e3w, e3b, h94, c0, cn);
  }

  // ---- ARMA layers ----
  auto arma = [&](const float* hin, int Fin, int Fout, int K,
                  const float* wi, const float* wr, const float* bias,
                  float* hout, int doTanh) {
    dim3 g((Fout + 63) / 64, (N_NODES + 63) / 64);
    for (int k = 0; k < K; ++k) {
      wmma_gemm<0, 0><<<g, 256, 0, stream>>>(hin, wi + (size_t)k * Fin * Fout,
                                             hinit + (size_t)k * N_NODES * Fout,
                                             (const float*)nullptr, N_NODES, Fout, Fin);
      wmma_gemm<0, 0><<<g, 256, 0, stream>>>(hin, wr + (size_t)k * Fin * Fout,
                                             root + (size_t)k * N_NODES * Fout,
                                             (const float*)nullptr, N_NODES, Fout, Fin);
    }
    hipMemsetAsync(agg, 0, (size_t)K * N_NODES * Fout * 4, stream);
    long nscat = (long)N_EDGES * Fout;
    scatter_k<<<(unsigned)((nscat + 255) / 256), 256, 0, stream>>>(hinit, src, dst, nrm,
                                                                   agg, N_EDGES, Fout, K);
    combine_k<<<(N_NODES * Fout + 255) / 256, 256, 0, stream>>>(agg, root, bias, hout,
                                                                Fout, K, doTanh);
  };

  arma(h94, 94, 128, 2, awi[0], awr[0], ab[0], hA, 1);
  arma(hA, 128, 128, 2, awi[1], awr[1], ab[1], hB, 1);
  arma(hB, 128, 128, 2, awi[2], awr[2], ab[2], hA, 1);
  arma(hA, 128, 128, 2, awi[3], awr[3], ab[3], hB, 1);
  arma(hB, 128, 128, 2, awi[4], awr[4], ab[4], hA, 1);
  arma(hA, 128, 16, 1, awi[5], awr[5], ab[5], hB, 0);

  softmax_k<<<(N_NODES + 255) / 256, 256, 0, stream>>>(hB, (float*)d_out, N_NODES, 16);
}
